// MultiTaskTuckerSoftShareLinear_82068235092812
// MI455X (gfx1250) — compile-verified, matrix operand order checked by reading, not run on hardware
//
#include <hip/hip_runtime.h>

// ---- sizes (fixed by the reference) ----
#define NT   8      // tasks
#define RR   64     // Tucker rank
#define NIN  4096
#define NOUT 4096
#define NB   1024   // batch

typedef __bf16 bf16;
typedef __bf16 bf16x16 __attribute__((ext_vector_type(16)));
typedef float  f32x8   __attribute__((ext_vector_type(8)));
typedef unsigned int u32x4 __attribute__((ext_vector_type(4)));
typedef int v4i __attribute__((ext_vector_type(4)));
typedef __attribute__((address_space(1))) v4i v4i_g;   // global <4 x i32>
typedef __attribute__((address_space(3))) v4i v4i_l;   // LDS    <4 x i32>

#if defined(__has_builtin)
#if __has_builtin(__builtin_amdgcn_global_load_async_to_lds_b128) && \
    __has_builtin(__builtin_amdgcn_s_wait_asynccnt)
#define USE_ASYNC_LDS 1
#endif
#endif
#ifndef USE_ASYNC_LDS
#define USE_ASYNC_LDS 0
#endif

// 16-byte global -> LDS copy (async gfx1250 path when available)
__device__ __forceinline__ void copy16_g2l(const bf16* g, bf16* l) {
#if USE_ASYNC_LDS
  __builtin_amdgcn_global_load_async_to_lds_b128((v4i_g*)g, (v4i_l*)l, 0, 0);
#else
  *(u32x4*)l = *(const u32x4*)g;
#endif
}
__device__ __forceinline__ void copy_fence() {
#if USE_ASYNC_LDS
  __builtin_amdgcn_s_wait_asynccnt(0);
#endif
}

// Load one 16x(K=32 window) bf16 WMMA fragment from an LDS tile whose rows are
// K-contiguous. Per ISA 7.12.2: lanes 0-15 hold K0-7 & K16-23, lanes 16-31
// hold K8-15 & K24-31 -> two ds_load_b128 per lane.
__device__ __forceinline__ bf16x16 ldsFrag(const bf16* base, int rowStride,
                                           int rowOff, int kOff, int lane) {
  const bf16* p = base + (rowOff + (lane & 15)) * rowStride + kOff + (lane >> 4) * 8;
  union { u32x4 v[2]; bf16x16 f; } u;
  u.v[0] = *(const u32x4*)p;
  u.v[1] = *(const u32x4*)(p + 16);
  return u.f;
}

__device__ __forceinline__ f32x8 wmma_bf16(bf16x16 a, bf16x16 b, f32x8 c) {
  return __builtin_amdgcn_wmma_f32_16x16x32_bf16(false, a, false, b,
                                                 (short)0, c, false, false);
}

__device__ __forceinline__ u32x4 pack8_bf16(float4 f0, float4 f1) {
  union { bf16 h[8]; u32x4 v; } pk;
  pk.h[0] = (bf16)f0.x; pk.h[1] = (bf16)f0.y; pk.h[2] = (bf16)f0.z; pk.h[3] = (bf16)f0.w;
  pk.h[4] = (bf16)f1.x; pk.h[5] = (bf16)f1.y; pk.h[6] = (bf16)f1.z; pk.h[7] = (bf16)f1.w;
  return pk.v;
}

// ---- kernel 1: Gt[t,k,l] = sum_j task[t,j] * G[j,k,l]  (8*64*64 outputs) ----
__global__ void k_taskcore(const float* __restrict__ task,
                           const float* __restrict__ G,
                           float* __restrict__ Gt) {
  int idx = blockIdx.x * blockDim.x + threadIdx.x;   // 32768 threads
  int l = idx & 63, k = (idx >> 6) & 63, t = idx >> 12;
  float acc = 0.f;
#pragma unroll 8
  for (int j = 0; j < RR; ++j) acc += task[t * RR + j] * G[(j * RR + k) * RR + l];
  Gt[idx] = acc;
}

// ---- kernel 2: WinT[t,l,i] = sum_k in_core[i,k] * Gt[t,k,l]  (bf16) ----
__global__ void k_win(const float* __restrict__ in_core,
                      const float* __restrict__ Gt,
                      bf16* __restrict__ WinT) {
  int idx = blockIdx.x * blockDim.x + threadIdx.x;   // 8*64*4096 threads
  int i = idx & 4095, l = (idx >> 12) & 63, t = idx >> 18;
  float acc = 0.f;
#pragma unroll 8
  for (int k = 0; k < RR; ++k) acc += in_core[i * RR + k] * Gt[(t * RR + k) * RR + l];
  WinT[idx] = (bf16)acc;   // idx == ((t*64+l)*4096 + i)
}

// ---- kernel 3: out_core f32 -> bf16 (already [o][l], K-contiguous N-major) ----
__global__ void k_cvtout(const float* __restrict__ oc, bf16* __restrict__ outT) {
  int idx = blockIdx.x * blockDim.x + threadIdx.x;
  outT[idx] = (bf16)oc[idx];
}

// ---- kernel 4 (stage 1): tmp[t,b,l] = x[:,t,:] @ Win[t]   M=1024 K=4096 N=64
// Block tile: 32M x 64N, BK=64, double-buffered LDS. 8 waves -> one 16x16 tile
// each, 2 WMMA/iter; next slab's async copies overlap current slab's WMMAs.
__global__ __launch_bounds__(256) void k_stage1(const float* __restrict__ x,
                                                const bf16* __restrict__ WinT,
                                                bf16* __restrict__ tmp) {
  __shared__ __align__(16) bf16 As[2][32 * 64];
  __shared__ __align__(16) bf16 Bs[2][64 * 64];
  const int t = blockIdx.z, mb = blockIdx.x, tid = threadIdx.x;
  const int lane = tid & 31, w = tid >> 5;
  const int mt = w & 1, nt = w >> 1;
  const int afr = tid >> 3, afc = (tid & 7) * 8;     // A fill: 32 rows x 64 f32
  const int bfr = tid >> 2, bfc = (tid & 3) * 16;    // B fill: 64 rows x 64 bf16
  const float* aG = x + ((size_t)(mb * 32 + afr) * NT + t) * NIN + afc;
  const bf16*  bG = WinT + ((size_t)t * RR + bfr) * NIN + bfc;
  f32x8 acc = {0, 0, 0, 0, 0, 0, 0, 0};

  // prologue: stage k0 = 0 into buffer 0
  {
    const float4 f0 = *(const float4*)(aG);
    const float4 f1 = *(const float4*)(aG + 4);
    *(u32x4*)&As[0][afr * 64 + afc] = pack8_bf16(f0, f1);
    copy16_g2l(bG,     &Bs[0][bfr * 64 + bfc]);
    copy16_g2l(bG + 8, &Bs[0][bfr * 64 + bfc + 8]);
    copy_fence();
    __syncthreads();
  }

  int buf = 0;
  for (int k0 = 64; k0 < NIN; k0 += 64) {
    const int nxt = buf ^ 1;
    // prefetch next slab: async B copy + A f32 global loads in flight
    const float4 f0 = *(const float4*)(aG + k0);
    const float4 f1 = *(const float4*)(aG + k0 + 4);
    copy16_g2l(bG + k0,     &Bs[nxt][bfr * 64 + bfc]);
    copy16_g2l(bG + k0 + 8, &Bs[nxt][bfr * 64 + bfc + 8]);
    // compute current slab while copies are in flight
    bf16x16 a0 = ldsFrag(As[buf], 64, mt * 16, 0,  lane);
    bf16x16 b0 = ldsFrag(Bs[buf], 64, nt * 16, 0,  lane);
    acc = wmma_bf16(a0, b0, acc);
    bf16x16 a1 = ldsFrag(As[buf], 64, mt * 16, 32, lane);
    bf16x16 b1 = ldsFrag(Bs[buf], 64, nt * 16, 32, lane);
    acc = wmma_bf16(a1, b1, acc);
    // convert + stage A into next buffer, then one barrier per iteration
    *(u32x4*)&As[nxt][afr * 64 + afc] = pack8_bf16(f0, f1);
    copy_fence();
    __syncthreads();
    buf = nxt;
  }

  // tail compute on last buffer
  {
    bf16x16 a0 = ldsFrag(As[buf], 64, mt * 16, 0,  lane);
    bf16x16 b0 = ldsFrag(Bs[buf], 64, nt * 16, 0,  lane);
    acc = wmma_bf16(a0, b0, acc);
    bf16x16 a1 = ldsFrag(As[buf], 64, mt * 16, 32, lane);
    bf16x16 b1 = ldsFrag(Bs[buf], 64, nt * 16, 32, lane);
    acc = wmma_bf16(a1, b1, acc);
  }

  // store C tile as bf16 into tmp[t][b][l]
  bf16* op = tmp + ((size_t)t * NB + mb * 32) * RR;
  const int col = nt * 16 + (lane & 15);
  const int rb  = mt * 16 + (lane >> 4) * 8;
#pragma unroll
  for (int r = 0; r < 8; ++r) op[(rb + r) * RR + col] = (bf16)acc[r];
}

// ---- kernel 5 (stage 2): y[b,t,o] = tmp[t] @ out_coreT  M=1024 K=64 N=4096
// Block tile: 64M x 128N, full K=64 resident. 8 waves x 4 tiles, 8 WMMA/wave.
__global__ __launch_bounds__(256) void k_stage2(const bf16* __restrict__ tmp,
                                                const bf16* __restrict__ outT,
                                                float* __restrict__ y) {
  __shared__ __align__(16) bf16 As[64 * 64];
  __shared__ __align__(16) bf16 Bs[128 * 64];
  const int t = blockIdx.z, mb = blockIdx.x, nb = blockIdx.y;
  const int tid = threadIdx.x, lane = tid & 31, w = tid >> 5;
  {
    const int fr = tid >> 2, fc = (tid & 3) * 16;    // 64 rows x 64 bf16
    const bf16* src = tmp + ((size_t)t * NB + mb * 64 + fr) * RR + fc;
    copy16_g2l(src,     &As[fr * 64 + fc]);
    copy16_g2l(src + 8, &As[fr * 64 + fc + 8]);
  }
  {
    const int fr = tid >> 1, fc = (tid & 1) * 32;    // 128 rows x 64 bf16
    const bf16* src = outT + (size_t)(nb * 128 + fr) * RR + fc;
    bf16* dst = &Bs[fr * 64 + fc];
    copy16_g2l(src,      dst);      copy16_g2l(src + 8,  dst + 8);
    copy16_g2l(src + 16, dst + 16); copy16_g2l(src + 24, dst + 24);
  }
  copy_fence();
  __syncthreads();

  const int mt = w & 3, ntB = (w >> 2) * 4;
  bf16x16 a0 = ldsFrag(As, 64, mt * 16, 0,  lane);
  bf16x16 a1 = ldsFrag(As, 64, mt * 16, 32, lane);
  const int rb = mb * 64 + mt * 16 + (lane >> 4) * 8;
#pragma unroll
  for (int j = 0; j < 4; ++j) {
    const int nt = ntB + j;
    f32x8 acc = {0, 0, 0, 0, 0, 0, 0, 0};
    bf16x16 b0 = ldsFrag(Bs, 64, nt * 16, 0,  lane);
    bf16x16 b1 = ldsFrag(Bs, 64, nt * 16, 32, lane);
    acc = wmma_bf16(a0, b0, acc);
    acc = wmma_bf16(a1, b1, acc);
    const int col = nb * 128 + nt * 16 + (lane & 15);
#pragma unroll
    for (int r = 0; r < 8; ++r)
      y[((size_t)(rb + r) * NT + t) * NOUT + col] = acc[r];
  }
}

extern "C" void kernel_launch(void* const* d_in, const int* in_sizes, int n_in,
                              void* d_out, int out_size, void* d_ws, size_t ws_size,
                              hipStream_t stream) {
  (void)in_sizes; (void)n_in; (void)out_size; (void)ws_size;
  const float* x      = (const float*)d_in[0];   // [1024,8,4096]
  const float* G      = (const float*)d_in[1];   // [64,64,64]
  const float* task   = (const float*)d_in[2];   // [8,64]
  const float* incore = (const float*)d_in[3];   // [4096,64]
  const float* ocore  = (const float*)d_in[4];   // [4096,64]
  float* y = (float*)d_out;                      // [1024,8,4096]

  char* ws = (char*)d_ws;                        // ~5.9 MB total
  float* Gt   = (float*)ws;                                   // 131072 B
  bf16*  WinT = (bf16*)(ws + 131072);                         // 4 MiB
  bf16*  outT = (bf16*)(ws + 131072 + 4194304);               // 512 KiB
  bf16*  tmp  = (bf16*)(ws + 131072 + 4194304 + 524288);      // 1 MiB

  k_taskcore<<<dim3(128),        dim3(256), 0, stream>>>(task, G, Gt);
  k_win     <<<dim3(8192),       dim3(256), 0, stream>>>(incore, Gt, WinT);
  k_cvtout  <<<dim3(1024),       dim3(256), 0, stream>>>(ocore, outT);
  k_stage1  <<<dim3(32, 1, 8),   dim3(256), 0, stream>>>(x, WinT, tmp);
  k_stage2  <<<dim3(16, 32, 8),  dim3(256), 0, stream>>>(tmp, outT, y);
}
